// TransformerBlock_25125558681997
// MI455X (gfx1250) — matrix operand order repaired and checked by reading, not compiled, not run on hardware
//
#include <hip/hip_runtime.h>
#include <hip/hip_bf16.h>

// ---------- types for CDNA5 WMMA ----------
typedef __attribute__((ext_vector_type(16))) __bf16         v16bf;
typedef __attribute__((ext_vector_type(16))) unsigned short v16u;
typedef __attribute__((ext_vector_type(8)))  float          v8f;

__device__ __forceinline__ unsigned short f2bf(float f) {
    unsigned int u = __float_as_uint(f);
    unsigned int r = u + 0x7FFFu + ((u >> 16) & 1u);   // round-to-nearest-even
    return (unsigned short)(r >> 16);
}

// ---------- fp32 -> bf16 (straight) ----------
__global__ __launch_bounds__(256) void k_f32_to_bf16(const float* __restrict__ in,
                                                     unsigned short* __restrict__ out,
                                                     long long n) {
    long long i = ((long long)blockIdx.x * blockDim.x + threadIdx.x) * 4;
#pragma unroll
    for (int j = 0; j < 4; ++j) {
        long long k = i + j;
        if (k < n) out[k] = f2bf(in[k]);
    }
}

// ---------- fp32 [R,C] -> bf16 [C,R] transposed convert (32x32 LDS tiles) ----------
__global__ __launch_bounds__(256) void k_f32_to_bf16_T(
    const float* __restrict__ in, unsigned short* __restrict__ out,
    int R, int C, long long sIn, long long sOut)
{
    __shared__ float tile[32][33];
    int z = blockIdx.z;
    const float* src = in + (long long)z * sIn;
    unsigned short* dst = out + (long long)z * sOut;
    int c0 = blockIdx.x * 32, r0 = blockIdx.y * 32;
    int tx = threadIdx.x & 31, ty0 = threadIdx.x >> 5;
#pragma unroll
    for (int i = 0; i < 4; ++i) {
        int ty = ty0 + i * 8;
        tile[ty][tx] = src[(long long)(r0 + ty) * C + c0 + tx];
    }
    __syncthreads();
#pragma unroll
    for (int i = 0; i < 4; ++i) {
        int ty = ty0 + i * 8;
        dst[(long long)(c0 + ty) * R + r0 + tx] = f2bf(tile[tx][ty]);
    }
}

// ---------- bf16 GEMM via v_wmma_f32_16x16x32_bf16 ----------
// C[M,N] = A[M,K] * B^T where B is stored [N,K] row-major (all operands pre-arranged).
// Block tile 128x128, K-step 32, 8 waves in 4x2 grid; each wave computes 32x64
// (2x4 WMMA 16x16 tiles -> 8 WMMAs per k-tile from 6 fragment loads).
// LDS tiles are stored fragment-contiguous: each lane's 16-bf16 WMMA operand is
// one 32-byte contiguous run -> 2x ds_load_b128, no repacking.
// Ping-pong double buffering overlaps global b128 loads with the WMMA pipe.
#define BM 128
#define BN 128
#define BK 32

__global__ __launch_bounds__(256) void k_gemm_bf16_wmma(
    const unsigned short* __restrict__ A, const unsigned short* __restrict__ B,
    const float* __restrict__ bias,
    float* __restrict__ outF, unsigned short* __restrict__ outH,
    int M, int N, int K,
    long long strideA, long long strideB, long long strideC,
    int relu)
{
    // A: 8 row-groups x 32 lanes x 16 elems = 4096 bf16 (8KB); B: 8 col-groups (8KB)
    __shared__ __align__(32) unsigned short AsF[2][8 * 32 * 16];
    __shared__ __align__(32) unsigned short BsF[2][8 * 32 * 16];

    const int z = blockIdx.z;
    const int tid  = threadIdx.x;
    const int wave = tid >> 5;
    const int lane = tid & 31;
    const int wm   = wave & 3;   // 4 wave-rows of 32
    const int wn   = wave >> 2;  // 2 wave-cols of 64
    const int m0   = blockIdx.y * BM;
    const int n0   = blockIdx.x * BN;

    // ---- per-thread staging addresses ----
    // A tile: 512 b128-chunks; thread handles chunks {tid, tid+256}.
    //   chunk c: m = c>>2, kq = (c&3)*8
    //   A frag layout: lane = ((kq>>3)&1)*16 + (m&15); elem base = ((kq>>4)&1)*8
    const int ca0 = tid, ca1 = tid + 256;
    const int am0 = ca0 >> 2, am1 = ca1 >> 2;
    const int aL0 = (((am0 >> 4) * 32 + ((ca0 & 1) << 4) + (am0 & 15)) << 4) + (((ca0 >> 1) & 1) << 3);
    const int aL1 = (((am1 >> 4) * 32 + ((ca1 & 1) << 4) + (am1 & 15)) << 4) + (((ca1 >> 1) & 1) << 3);
    // B tile: 512 b128-chunks; thread handles chunks {tid, tid+256}.
    //   chunk c: n = c>>2, kq = (c&3)*8
    //   B frag layout: lane = ((kq>>3)&1)*16 + (n&15); elem base = ((kq>>4)&1)*8
    const int cb0 = tid, cb1 = tid + 256;
    const int bn0 = cb0 >> 2, bn1 = cb1 >> 2;
    const int bL0 = (((bn0 >> 4) * 32 + ((cb0 & 1) << 4) + (bn0 & 15)) << 4) + (((cb0 >> 1) & 1) << 3);
    const int bL1 = (((bn1 >> 4) * 32 + ((cb1 & 1) << 4) + (bn1 & 15)) << 4) + (((cb1 >> 1) & 1) << 3);

    const unsigned short* aG0 = A + (long long)z * strideA + (long long)(m0 + am0) * K + ((ca0 & 3) << 3);
    const unsigned short* aG1 = A + (long long)z * strideA + (long long)(m0 + am1) * K + ((ca1 & 3) << 3);
    const unsigned short* bG0 = B + (long long)z * strideB + (long long)(n0 + bn0) * K + ((cb0 & 3) << 3);
    const unsigned short* bG1 = B + (long long)z * strideB + (long long)(n0 + bn1) * K + ((cb1 & 3) << 3);

    v8f acc[2][4] = {};

    // ---- prologue: stage tile 0 into buffer 0 ----
    uint4 ra0 = *(const uint4*)aG0;
    uint4 ra1 = *(const uint4*)aG1;
    uint4 rb0 = *(const uint4*)bG0;
    uint4 rb1 = *(const uint4*)bG1;
    *(uint4*)&AsF[0][aL0] = ra0;
    *(uint4*)&AsF[0][aL1] = ra1;
    *(uint4*)&BsF[0][bL0] = rb0;
    *(uint4*)&BsF[0][bL1] = rb1;

    const int nT = K / BK;
    for (int t = 0; t < nT; ++t) {
        __syncthreads();                       // buf[t&1] visible to all waves
        const bool more = (t + 1 < nT);
        if (more) {
            aG0 += BK; aG1 += BK; bG0 += BK; bG1 += BK;
            ra0 = *(const uint4*)aG0;
            ra1 = *(const uint4*)aG1;
            rb0 = *(const uint4*)bG0;
            rb1 = *(const uint4*)bG1;
            if (t + 2 < nT) {                  // gfx1250 global_prefetch path
                __builtin_prefetch(aG0 + BK, 0, 1);
                __builtin_prefetch(bG0 + BK, 0, 1);
            }
        }
        const int cur = t & 1;

        v16u afr[2], bfr[4];
#pragma unroll
        for (int i = 0; i < 2; ++i)
            afr[i] = *(const v16u*)&AsF[cur][(((wm * 2 + i) * 32 + lane) << 4)];
#pragma unroll
        for (int j = 0; j < 4; ++j)
            bfr[j] = *(const v16u*)&BsF[cur][(((wn * 4 + j) * 32 + lane) << 4)];

#pragma unroll
        for (int i = 0; i < 2; ++i)
#pragma unroll
            for (int j = 0; j < 4; ++j)
                acc[i][j] = __builtin_amdgcn_wmma_f32_16x16x32_bf16(
                    false, __builtin_bit_cast(v16bf, afr[i]),
                    false, __builtin_bit_cast(v16bf, bfr[j]),
                    (short)0, acc[i][j], false, false);

        if (more) {
            const int nx = cur ^ 1;
            *(uint4*)&AsF[nx][aL0] = ra0;
            *(uint4*)&AsF[nx][aL1] = ra1;
            *(uint4*)&BsF[nx][bL0] = rb0;
            *(uint4*)&BsF[nx][bL1] = rb1;
        }
    }

    // ---- epilogue: bias / relu / fp32 and/or bf16 stores ----
    const int hl  = lane >> 4;
    const int l16 = lane & 15;
    float* oF = outF ? outF + (long long)z * strideC : nullptr;
    unsigned short* oH = outH ? outH + (long long)z * strideC : nullptr;
#pragma unroll
    for (int i = 0; i < 2; ++i)
#pragma unroll
        for (int j = 0; j < 4; ++j) {
            int nb = n0 + wn * 64 + j * 16 + l16;
            float bv = bias ? bias[nb] : 0.f;
#pragma unroll
            for (int r = 0; r < 8; ++r) {
                int m = m0 + wm * 32 + i * 16 + hl * 8 + r;
                float v = acc[i][j][r] + bv;
                if (relu) v = fmaxf(v, 0.f);
                long long o = (long long)m * N + nb;
                if (oF) oF[o] = v;
                if (oH) oH[o] = f2bf(v);
            }
        }
}

// ---------- row softmax (scaled), fp32 in -> bf16 out, cols <= 1024 ----------
__global__ __launch_bounds__(256) void k_softmax_bf16(const float* __restrict__ E,
                                                      unsigned short* __restrict__ O,
                                                      int cols, float scale)
{
    int row = blockIdx.x;
    const float* e = E + (long long)row * cols;
    unsigned short* o = O + (long long)row * cols;
    __shared__ float red[256];
    __shared__ float ex[1024];

    float vmax = -3.402823466e38f;
    for (int c = threadIdx.x; c < cols; c += 256) vmax = fmaxf(vmax, e[c]);
    red[threadIdx.x] = vmax; __syncthreads();
    for (int st = 128; st > 0; st >>= 1) {
        if (threadIdx.x < st) red[threadIdx.x] = fmaxf(red[threadIdx.x], red[threadIdx.x + st]);
        __syncthreads();
    }
    float m = red[0] * scale; __syncthreads();

    float s = 0.f;
    for (int c = threadIdx.x; c < cols; c += 256) {
        float t = __expf(e[c] * scale - m);
        ex[c] = t; s += t;
    }
    red[threadIdx.x] = s; __syncthreads();
    for (int st = 128; st > 0; st >>= 1) {
        if (threadIdx.x < st) red[threadIdx.x] += red[threadIdx.x + st];
        __syncthreads();
    }
    float inv = 1.f / red[0]; __syncthreads();
    for (int c = threadIdx.x; c < cols; c += 256) o[c] = f2bf(ex[c] * inv);
}

// ---------- layernorm over concat(A_row[da], B_row[db]), D <= 1536 ----------
__global__ __launch_bounds__(256) void k_ln_rows(
    const float* __restrict__ A, int da, const float* __restrict__ B, int db,
    const float* __restrict__ g, const float* __restrict__ bt,
    float* __restrict__ outF, unsigned short* __restrict__ outH)
{
    int row = blockIdx.x;
    int D = da + db;
    __shared__ float red[256];
    float v[6];

    float s = 0.f;
#pragma unroll
    for (int i = 0; i < 6; ++i) {
        int c = threadIdx.x + i * 256;
        float x = 0.f;
        if (c < D) x = (c < da) ? A[(long long)row * da + c]
                                : B[(long long)row * db + (c - da)];
        v[i] = x; s += x;
    }
    red[threadIdx.x] = s; __syncthreads();
    for (int st = 128; st > 0; st >>= 1) {
        if (threadIdx.x < st) red[threadIdx.x] += red[threadIdx.x + st];
        __syncthreads();
    }
    float mu = red[0] / (float)D; __syncthreads();

    float s2 = 0.f;
#pragma unroll
    for (int i = 0; i < 6; ++i) {
        int c = threadIdx.x + i * 256;
        if (c < D) { float d = v[i] - mu; s2 += d * d; }
    }
    red[threadIdx.x] = s2; __syncthreads();
    for (int st = 128; st > 0; st >>= 1) {
        if (threadIdx.x < st) red[threadIdx.x] += red[threadIdx.x + st];
        __syncthreads();
    }
    float rstd = rsqrtf(red[0] / (float)D + 1e-5f); __syncthreads();

#pragma unroll
    for (int i = 0; i < 6; ++i) {
        int c = threadIdx.x + i * 256;
        if (c < D) {
            float y = (v[i] - mu) * rstd * g[c] + bt[c];
            long long o = (long long)row * D + c;
            if (outF) outF[o] = y;
            if (outH) outH[o] = f2bf(y);
        }
    }
}

// ---------- host orchestration ----------
extern "C" void kernel_launch(void* const* d_in, const int* in_sizes, int n_in,
                              void* d_out, int out_size, void* d_ws, size_t ws_size,
                              hipStream_t stream) {
    (void)in_sizes; (void)n_in; (void)out_size; (void)ws_size;

    const float* value = (const float*)d_in[0];   // [16,1024,512]
    const float* key   = (const float*)d_in[1];   // [16,1024,512]
    const float* query = (const float*)d_in[2];   // [16,1024,512]
    const float* Wk    = (const float*)d_in[3];   // [512,512]
    const float* Wq    = (const float*)d_in[4];   // [512,512]
    const float* Wo    = (const float*)d_in[5];   // [512,512]
    const float* bo    = (const float*)d_in[6];   // [512]
    const float* ga    = (const float*)d_in[7];   // [1024]
    const float* ba    = (const float*)d_in[8];
    const float* g1    = (const float*)d_in[9];   // [1536]
    const float* b1    = (const float*)d_in[10];
    const float* W1    = (const float*)d_in[11];  // [1536,3072]
    const float* bf1   = (const float*)d_in[12];  // [3072]
    const float* W2    = (const float*)d_in[13];  // [3072,1536]
    const float* bf2   = (const float*)d_in[14];  // [1536]
    const float* g2    = (const float*)d_in[15];
    const float* b2    = (const float*)d_in[16];

    const long long NB = 16, LQ = 1024, LK = 1024, DQ = 512, DK = 512, DP = 512, DE = 512;
    const long long D2 = 1536, DH = 3072;
    const long long ROWS = NB * LQ;               // 16384

    // ---- carve workspace ----
    size_t off = 0;
    auto alloc = [&](size_t bytes) -> void* {
        void* p = (char*)d_ws + off;
        off += (bytes + 255) & ~(size_t)255;
        return p;
    };
    unsigned short* value_t = (unsigned short*)alloc(NB * DK * LK * 2);  // [n, DK, LK]
    unsigned short* key_h   = (unsigned short*)alloc(NB * LK * DK * 2);
    unsigned short* query_h = (unsigned short*)alloc(NB * LQ * DQ * 2);
    unsigned short* Wk_t    = (unsigned short*)alloc(DP * DK * 2);
    unsigned short* Wq_t    = (unsigned short*)alloc(DP * DQ * 2);
    unsigned short* Wo_t    = (unsigned short*)alloc(DE * DK * 2);
    unsigned short* W1_t    = (unsigned short*)alloc(DH * D2 * 2);
    unsigned short* W2_t    = (unsigned short*)alloc(D2 * DH * 2);
    unsigned short* kp_h    = (unsigned short*)alloc(NB * LK * DP * 2);
    unsigned short* qp_h    = (unsigned short*)alloc(NB * LQ * DP * 2);
    float*          energy  = (float*)alloc(NB * LQ * LK * 4);           // reused as attnout
    unsigned short* attn_h  = (unsigned short*)alloc(NB * LQ * LK * 2);  // reused as outWo (f32, same bytes)
    unsigned short* xln_h   = (unsigned short*)alloc(ROWS * D2 * 2);
    unsigned short* h_h     = (unsigned short*)alloc(ROWS * DH * 2);

    unsigned short* av_h    = key_h;            // key bf16 dead after kp GEMM
    float*          outWo_f = (float*)attn_h;   // attn dead after attn*V GEMM
    float*          attnout = energy;           // energy dead after softmax
    float*          out_f   = (float*)d_out;    // FFN2 output, LN'd in place

    // ---- 1) conversions: A-side straight, B-side transposed to [N,K] ----
    auto cvt = [&](const float* s, unsigned short* d, long long n) {
        k_f32_to_bf16<<<dim3((unsigned)((n + 1023) / 1024)), 256, 0, stream>>>(s, d, n);
    };
    auto cvtT = [&](const float* s, unsigned short* d, int R, int C, int batch,
                    long long sIn, long long sOut) {
        k_f32_to_bf16_T<<<dim3(C / 32, R / 32, batch), 256, 0, stream>>>(s, d, R, C, sIn, sOut);
    };
    cvt(key,   key_h,   NB * LK * DK);
    cvt(query, query_h, NB * LQ * DQ);
    cvtT(value, value_t, (int)LK, (int)DK, (int)NB, LK * DK, DK * LK);   // [n,LK,DK]->[n,DK,LK]
    cvtT(Wk, Wk_t, (int)DK, (int)DP, 1, 0, 0);                           // [DK,DP]->[DP,DK]
    cvtT(Wq, Wq_t, (int)DQ, (int)DP, 1, 0, 0);
    cvtT(Wo, Wo_t, (int)DK, (int)DE, 1, 0, 0);
    cvtT(W1, W1_t, (int)D2, (int)DH, 1, 0, 0);                           // [1536,3072]->[3072,1536]
    cvtT(W2, W2_t, (int)DH, (int)D2, 1, 0, 0);

    // ---- 2) kp = key @ Wk (bf16 out, [16384,512]) ----
    k_gemm_bf16_wmma<<<dim3(DP / BN, ROWS / BM, 1), 256, 0, stream>>>(
        key_h, Wk_t, nullptr, nullptr, kp_h,
        (int)ROWS, (int)DP, (int)DK, 0, 0, 0, 0);
    // ---- 3) qp = query @ Wq ----
    k_gemm_bf16_wmma<<<dim3(DP / BN, ROWS / BM, 1), 256, 0, stream>>>(
        query_h, Wq_t, nullptr, nullptr, qp_h,
        (int)ROWS, (int)DP, (int)DQ, 0, 0, 0, 0);
    // ---- 4) energy = qp @ kp^T (batched, fp32 out); kp is already [N=LK, K=DP] ----
    k_gemm_bf16_wmma<<<dim3(LK / BN, LQ / BM, NB), 256, 0, stream>>>(
        qp_h, kp_h, nullptr, energy, nullptr,
        (int)LQ, (int)LK, (int)DP, LQ * DP, LK * DP, LQ * LK, 0);
    // ---- 5) softmax(energy / sqrt(DP)) -> attn (bf16) ----
    k_softmax_bf16<<<dim3((unsigned)(NB * LQ)), 256, 0, stream>>>(
        energy, attn_h, (int)LK, 0.044194173824159216f /* 1/sqrt(512) */);
    // ---- 6) av = attn @ value (batched, bf16 out); value_t is [N=DK, K=LK] ----
    k_gemm_bf16_wmma<<<dim3(DK / BN, LQ / BM, NB), 256, 0, stream>>>(
        attn_h, value_t, nullptr, nullptr, av_h,
        (int)LQ, (int)DK, (int)LK, LQ * LK, DK * LK, LQ * DK, 0);
    // ---- 7) outWo = av @ Wo + bo (fp32 out) ----
    k_gemm_bf16_wmma<<<dim3(DE / BN, ROWS / BM, 1), 256, 0, stream>>>(
        av_h, Wo_t, bo, outWo_f, nullptr,
        (int)ROWS, (int)DE, (int)DK, 0, 0, 0, 0);
    // ---- 8) attn_out = LN(concat(outWo, query); ga, ba) (fp32) ----
    k_ln_rows<<<dim3((unsigned)ROWS), 256, 0, stream>>>(
        outWo_f, (int)DE, query, (int)DQ, ga, ba, attnout, nullptr);
    // ---- 9) x = LN(concat(attn_out, query); g1, b1) (bf16) ----
    k_ln_rows<<<dim3((unsigned)ROWS), 256, 0, stream>>>(
        attnout, (int)(DE + DQ), query, (int)DQ, g1, b1, nullptr, xln_h);
    // ---- 10) h = relu(x @ W1 + bf1) (bf16) ----
    k_gemm_bf16_wmma<<<dim3(DH / BN, ROWS / BM, 1), 256, 0, stream>>>(
        xln_h, W1_t, bf1, nullptr, h_h,
        (int)ROWS, (int)DH, (int)D2, 0, 0, 0, 1);
    // ---- 11) f = h @ W2 + bf2 (fp32 -> d_out) ----
    k_gemm_bf16_wmma<<<dim3(D2 / BN, ROWS / BM, 1), 256, 0, stream>>>(
        h_h, W2_t, bf2, out_f, nullptr,
        (int)ROWS, (int)D2, (int)DH, 0, 0, 0, 0);
    // ---- 12) out = LN(f; g2, b2), in place on d_out ----
    k_ln_rows<<<dim3((unsigned)ROWS), 256, 0, stream>>>(
        out_f, (int)D2, nullptr, 0, g2, b2, out_f, nullptr);
}
